// MultiHeadAttention_84696755077830
// MI455X (gfx1250) — matrix operand order repaired and checked by reading, not compiled, not run on hardware
//
#include <hip/hip_runtime.h>

typedef _Float16 half_t;
typedef __attribute__((ext_vector_type(16))) _Float16 v16h;
typedef __attribute__((ext_vector_type(8)))  _Float16 v8h;
typedef __attribute__((ext_vector_type(4)))  _Float16 v4h;
typedef __attribute__((ext_vector_type(8)))  float    v8f;
typedef __attribute__((ext_vector_type(4)))  float    v4f;

#define D_MODEL 1024
#define NHEAD   16
#define HEAD_DIM 64
#define BATCH   2
#define SEQ     2048

#define INFF (__builtin_inff())

union V16H { v16h v; v8h h[2]; };

// A fragment (16x32 f16, M x K): lane = row (lane&15), half h=lane>>4 selects
// k = {8h..8h+7} and {16+8h..16+8h+7}.  Source: row-major [row][k].
__device__ __forceinline__ v16h load_a_frag(const half_t* base, int stride, int lane) {
  const int m = lane & 15, hh = lane >> 4;
  const half_t* p = base + m * stride + 8 * hh;
  V16H r;
  r.h[0] = *(const v8h*)(p);
  r.h[1] = *(const v8h*)(p + 16);
  return r.v;
}

// B fragment (32x16 f16, K x N): lane = column (lane&15), half h selects
// k = {8h..8h+7} and {16+8h..16+8h+7}.  Source: row-major [col][k] (B transposed).
__device__ __forceinline__ v16h load_b_frag(const half_t* base, int stride, int lane) {
  const int n = lane & 15, hh = lane >> 4;
  const half_t* p = base + n * stride + 8 * hh;
  V16H r;
  r.h[0] = *(const v8h*)(p);
  r.h[1] = *(const v8h*)(p + 16);
  return r.v;
}

__device__ __forceinline__ v8f wmma_f16(v16h a, v16h b, v8f c) {
  return __builtin_amdgcn_wmma_f32_16x16x32_f16(false, a, false, b, (short)0, c,
                                                false, false);
}

// CDNA5 async global->LDS copy (16B), tracked by ASYNCcnt.
__device__ __forceinline__ void async_copy_b128(half_t* lds_dst, const half_t* gsrc) {
  unsigned loff = (unsigned)(uintptr_t)lds_dst;  // LDS aperture: low 32 bits = LDS addr
  asm volatile("global_load_async_to_lds_b128 %0, %1, off"
               :: "v"(loff), "v"(gsrc) : "memory");
}
__device__ __forceinline__ void async_wait0() {
  asm volatile("s_wait_asynccnt 0x0" ::: "memory");
}

// ---------------------------------------------------------------------------
// GEMM: C[M,N] = X[M,K]*W[K,N] + bias.  256 thr = 8 waves (2x4), 128x256 tile.
// LDS ping-pong double buffering: stage tile t+1 while WMMAs consume tile t.
// ---------------------------------------------------------------------------
#define GT_M 128
#define GT_N 256
#define GT_K 32
#define LDA  40
#define LDB  40
#define NKT  (D_MODEL / GT_K)   // 32 k-tiles

__device__ __forceinline__ void stage_a_f32(half_t* sA, const float* __restrict__ X,
                                            int m0, int k0, int tid) {
#pragma unroll
  for (int j = 0; j < 4; ++j) {
    int i = (tid + j * 256) * 4;
    int m = i >> 5, k = i & 31;
    v4f x = *(const v4f*)(X + (size_t)(m0 + m) * D_MODEL + k0 + k);
    v4h y;
    y[0] = (half_t)x[0]; y[1] = (half_t)x[1];
    y[2] = (half_t)x[2]; y[3] = (half_t)x[3];
    *(v4h*)(sA + m * LDA + k) = y;
  }
}
__device__ __forceinline__ void stage_a_f16(half_t* sA, const half_t* __restrict__ X,
                                            int m0, int k0, int tid) {
#pragma unroll
  for (int j = 0; j < 2; ++j) {
    int i = tid + j * 256;
    int m = i >> 2, seg = (i & 3) * 8;
    *(v8h*)(sA + m * LDA + seg) =
        *(const v8h*)(X + (size_t)(m0 + m) * D_MODEL + k0 + seg);
  }
}
__device__ __forceinline__ void stage_b_f32(half_t* sB, const float* __restrict__ W,
                                            int n0, int k0, int tid) {
#pragma unroll
  for (int j = 0; j < 8; ++j) {
    int i = (tid + j * 256) * 4;
    int n = i & 255, k = i >> 8;
    v4f x = *(const v4f*)(W + (size_t)(k0 + k) * D_MODEL + n0 + n);
    sB[(n + 0) * LDB + k] = (half_t)x[0];
    sB[(n + 1) * LDB + k] = (half_t)x[1];
    sB[(n + 2) * LDB + k] = (half_t)x[2];
    sB[(n + 3) * LDB + k] = (half_t)x[3];
  }
}

// OUT_MODE 0: f16 out [B,H,S,Dh] (Q,K).  OUT_MODE 1: f16 out [B,H,Dh,S] (V).
template <int OUT_MODE>
__global__ __launch_bounds__(256) void gemm_qkv_kernel(
    const float* __restrict__ X, const float* __restrict__ W,
    const float* __restrict__ bias, half_t* __restrict__ out) {
  __shared__ __align__(16) half_t sA[2][GT_M * LDA];
  __shared__ __align__(16) half_t sB[2][GT_N * LDB];
  const int tid = threadIdx.x;
  const int lane = tid & 31, wid = tid >> 5;
  const int wm = wid & 1, wn = wid >> 1;
  const int m0 = blockIdx.x * GT_M;
  const int n0 = blockIdx.y * GT_N;

  v8f acc[4][4] = {};
  stage_a_f32(sA[0], X, m0, 0, tid);
  stage_b_f32(sB[0], W, n0, 0, tid);
  __syncthreads();
  for (int kt = 0; kt < NKT; ++kt) {
    const int cur = kt & 1;
    if (kt + 1 < NKT) {  // stage next tile; overlaps with WMMAs below
      stage_a_f32(sA[cur ^ 1], X, m0, (kt + 1) * GT_K, tid);
      stage_b_f32(sB[cur ^ 1], W, n0, (kt + 1) * GT_K, tid);
      if (kt + 2 < NKT) {
        __builtin_prefetch(X + (size_t)(m0 + (tid >> 1)) * D_MODEL + (kt + 2) * GT_K, 0, 0);
        __builtin_prefetch(W + (size_t)((kt + 2) * GT_K + (tid >> 3)) * D_MODEL + n0, 0, 0);
      }
    }
    v16h af[4], bf[4];
#pragma unroll
    for (int mi = 0; mi < 4; ++mi)
      af[mi] = load_a_frag(sA[cur] + (wm * 64 + mi * 16) * LDA, LDA, lane);
#pragma unroll
    for (int ni = 0; ni < 4; ++ni)
      bf[ni] = load_b_frag(sB[cur] + (wn * 64 + ni * 16) * LDB, LDB, lane);
#pragma unroll
    for (int mi = 0; mi < 4; ++mi)
#pragma unroll
      for (int ni = 0; ni < 4; ++ni)
        acc[mi][ni] = wmma_f16(af[mi], bf[ni], acc[mi][ni]);
    __syncthreads();
  }
  const int hh = lane >> 4;
#pragma unroll
  for (int mi = 0; mi < 4; ++mi) {
#pragma unroll
    for (int ni = 0; ni < 4; ++ni) {
      int gn = n0 + wn * 64 + ni * 16 + (lane & 15);
      float bv = bias[gn];
      int head = gn >> 6, d = gn & 63;
      int gmb = m0 + wm * 64 + mi * 16 + 8 * hh;
      if (OUT_MODE == 0) {
#pragma unroll
        for (int r = 0; r < 8; ++r) {
          int gm = gmb + r;
          int bb = gm >> 11, ss = gm & (SEQ - 1);
          out[((size_t)(bb * NHEAD + head) * SEQ + ss) * HEAD_DIM + d] =
              (half_t)(acc[mi][ni][r] + bv);
        }
      } else {
        int bb = gmb >> 11, s0 = gmb & (SEQ - 1);
        v8h y;
#pragma unroll
        for (int r = 0; r < 8; ++r) y[r] = (half_t)(acc[mi][ni][r] + bv);
        *(v8h*)(out + ((size_t)(bb * NHEAD + head) * HEAD_DIM + d) * SEQ + s0) = y;
      }
    }
  }
}

__global__ __launch_bounds__(256) void gemm_out_kernel(
    const half_t* __restrict__ X, const float* __restrict__ W,
    const float* __restrict__ bias, float* __restrict__ out /*[M,N] f32*/) {
  __shared__ __align__(16) half_t sA[2][GT_M * LDA];
  __shared__ __align__(16) half_t sB[2][GT_N * LDB];
  const int tid = threadIdx.x;
  const int lane = tid & 31, wid = tid >> 5;
  const int wm = wid & 1, wn = wid >> 1;
  const int m0 = blockIdx.x * GT_M;
  const int n0 = blockIdx.y * GT_N;

  v8f acc[4][4] = {};
  stage_a_f16(sA[0], X, m0, 0, tid);
  stage_b_f32(sB[0], W, n0, 0, tid);
  __syncthreads();
  for (int kt = 0; kt < NKT; ++kt) {
    const int cur = kt & 1;
    if (kt + 1 < NKT) {
      stage_a_f16(sA[cur ^ 1], X, m0, (kt + 1) * GT_K, tid);
      stage_b_f32(sB[cur ^ 1], W, n0, (kt + 1) * GT_K, tid);
      if (kt + 2 < NKT) {
        __builtin_prefetch(X + (size_t)(m0 + (tid >> 1)) * D_MODEL + (kt + 2) * GT_K, 0, 0);
        __builtin_prefetch(W + (size_t)((kt + 2) * GT_K + (tid >> 3)) * D_MODEL + n0, 0, 0);
      }
    }
    v16h af[4], bf[4];
#pragma unroll
    for (int mi = 0; mi < 4; ++mi)
      af[mi] = load_a_frag(sA[cur] + (wm * 64 + mi * 16) * LDA, LDA, lane);
#pragma unroll
    for (int ni = 0; ni < 4; ++ni)
      bf[ni] = load_b_frag(sB[cur] + (wn * 64 + ni * 16) * LDB, LDB, lane);
#pragma unroll
    for (int mi = 0; mi < 4; ++mi)
#pragma unroll
      for (int ni = 0; ni < 4; ++ni)
        acc[mi][ni] = wmma_f16(af[mi], bf[ni], acc[mi][ni]);
    __syncthreads();
  }
  const int hh = lane >> 4;
#pragma unroll
  for (int mi = 0; mi < 4; ++mi) {
#pragma unroll
    for (int ni = 0; ni < 4; ++ni) {
      int gn = n0 + wn * 64 + ni * 16 + (lane & 15);
      float bv = bias[gn];
#pragma unroll
      for (int r = 0; r < 8; ++r) {
        int gm = m0 + wm * 64 + mi * 16 + r + 8 * hh;
        out[(size_t)gm * D_MODEL + gn] = acc[mi][ni][r] + bv;
      }
    }
  }
}

// ---------------------------------------------------------------------------
// Flash attention (causal).  256 thr = 8 waves, 16 q-rows/wave.
// K [B,H,S,Dh]; V transposed [B,H,Dh,S].  Double-buffered 64-key K/V LDS tiles
// staged with CDNA5 global_load_async_to_lds_b128 (ASYNCcnt pipelining).
// ---------------------------------------------------------------------------
#define KV_TILE 64
#define LDK 72
#define LDP 40

__device__ __forceinline__ void stage_kv_async(half_t* sK, half_t* sV,
                                               const half_t* Kb, const half_t* VTb,
                                               int kv0, int tid) {
#pragma unroll
  for (int j = 0; j < 2; ++j) {
    int i = tid + j * 256;  // 0..511
    int row = i >> 3, seg = (i & 7) * 8;
    async_copy_b128(sK + row * LDK + seg,
                    Kb + (size_t)(kv0 + row) * HEAD_DIM + seg);
    async_copy_b128(sV + row * LDK + seg,
                    VTb + (size_t)row * SEQ + kv0 + seg);
  }
}

__global__ __launch_bounds__(256) void attn_kernel(
    const half_t* __restrict__ Q, const half_t* __restrict__ K,
    const half_t* __restrict__ VT, half_t* __restrict__ CTX /*[B,S,D] f16*/) {
  __shared__ __align__(16) half_t sK[2][KV_TILE * LDK];    // [key][d]
  __shared__ __align__(16) half_t sV[2][HEAD_DIM * LDK];   // [d][key]
  __shared__ __align__(16) half_t sP[8 * 16 * LDP];        // per-wave 16x32 scratch
  const int tid = threadIdx.x, lane = tid & 31, wid = tid >> 5;
  const int b = blockIdx.z, head = blockIdx.y;
  const int q0 = blockIdx.x * 128 + wid * 16;
  const int qmaxWG = blockIdx.x * 128 + 127;
  const int nTiles = (qmaxWG >> 6) + 1;  // WG-uniform
  const size_t bh = ((size_t)b * NHEAD + head);
  const half_t* Qb  = Q  + bh * SEQ * HEAD_DIM;
  const half_t* Kb  = K  + bh * SEQ * HEAD_DIM;
  const half_t* VTb = VT + bh * HEAD_DIM * SEQ;

  v16h qa0 = load_a_frag(Qb + (size_t)q0 * HEAD_DIM, HEAD_DIM, lane);
  v16h qa1 = load_a_frag(Qb + (size_t)q0 * HEAD_DIM + 32, HEAD_DIM, lane);

  v8f acc[4] = {};
  float mrow[8], lrow[8];
#pragma unroll
  for (int r = 0; r < 8; ++r) { mrow[r] = -INFF; lrow[r] = 0.f; }

  half_t* pw = sP + wid * 16 * LDP;
  const int col = lane & 15;
  const int rbase = (lane >> 4) * 8;

  stage_kv_async(sK[0], sV[0], Kb, VTb, 0, tid);
  async_wait0();
  __syncthreads();

  for (int t = 0; t < nTiles; ++t) {
    const int cur = t & 1;
    const int kv0 = t * KV_TILE;
    if (t + 1 < nTiles)  // background prefetch of next K/V tile (ASYNCcnt)
      stage_kv_async(sK[cur ^ 1], sV[cur ^ 1], Kb, VTb, kv0 + KV_TILE, tid);
#pragma unroll
    for (int sub = 0; sub < 2; ++sub) {
      const int k0 = kv0 + sub * 32;
      if (k0 <= q0 + 15) {  // wave-uniform
        const half_t* kbase = sK[cur] + (k0 - kv0) * LDK;
        v8f s0 = {}, s1 = {};
        s0 = wmma_f16(qa0, load_b_frag(kbase, LDK, lane), s0);
        s0 = wmma_f16(qa1, load_b_frag(kbase + 32, LDK, lane), s0);
        s1 = wmma_f16(qa0, load_b_frag(kbase + 16 * LDK, LDK, lane), s1);
        s1 = wmma_f16(qa1, load_b_frag(kbase + 16 * LDK + 32, LDK, lane), s1);

        float x0[8], x1[8];
        const int kc0 = k0 + col, kc1 = k0 + 16 + col;
#pragma unroll
        for (int r = 0; r < 8; ++r) {
          int qrow = q0 + rbase + r;
          float a0 = s0[r] * 0.125f;
          float a1 = s1[r] * 0.125f;
          if (kc0 > qrow) a0 = -INFF;
          if (kc1 > qrow) a1 = -INFF;
          x0[r] = a0; x1[r] = a1;
        }
        float scale[8];
#pragma unroll
        for (int r = 0; r < 8; ++r) {
          float tmax = fmaxf(x0[r], x1[r]);
#pragma unroll
          for (int off = 1; off < 16; off <<= 1)
            tmax = fmaxf(tmax, __shfl_xor(tmax, off, 32));
          float mn = fmaxf(mrow[r], tmax);
          float p0 = __expf(x0[r] - mn);
          float p1 = __expf(x1[r] - mn);
          float rs = p0 + p1;
#pragma unroll
          for (int off = 1; off < 16; off <<= 1)
            rs += __shfl_xor(rs, off, 32);
          float sc = __expf(mrow[r] - mn);
          lrow[r] = lrow[r] * sc + rs;
          mrow[r] = mn;
          scale[r] = sc;
          pw[(rbase + r) * LDP + col] = (half_t)p0;
          pw[(rbase + r) * LDP + 16 + col] = (half_t)p1;
        }
#pragma unroll
        for (int f = 0; f < 4; ++f)
#pragma unroll
          for (int r = 0; r < 8; ++r) acc[f][r] *= scale[r];

        v16h pa = load_a_frag(pw, LDP, lane);
#pragma unroll
        for (int f = 0; f < 4; ++f)
          acc[f] = wmma_f16(
              pa, load_b_frag(sV[cur] + f * 16 * LDK + (k0 - kv0), LDK, lane), acc[f]);
      }
    }
    async_wait0();   // next tile's async copies landed
    __syncthreads();
  }

  float inv[8];
#pragma unroll
  for (int r = 0; r < 8; ++r) inv[r] = 1.0f / lrow[r];
#pragma unroll
  for (int f = 0; f < 4; ++f) {
    int d = head * HEAD_DIM + f * 16 + col;
#pragma unroll
    for (int r = 0; r < 8; ++r) {
      int row = q0 + rbase + r;
      CTX[((size_t)b * SEQ + row) * D_MODEL + d] = (half_t)(acc[f][r] * inv[r]);
    }
  }
}

// ---------------------------------------------------------------------------
extern "C" void kernel_launch(void* const* d_in, const int* in_sizes, int n_in,
                              void* d_out, int out_size, void* d_ws, size_t ws_size,
                              hipStream_t stream) {
  const float* query = (const float*)d_in[0];
  const float* key   = (const float*)d_in[1];
  const float* value = (const float*)d_in[2];
  // d_in[3] = attn_mask (causal) — implemented analytically in attn_kernel
  const float* Wq = (const float*)d_in[4];
  const float* bq = (const float*)d_in[5];
  const float* Wk = (const float*)d_in[6];
  const float* bk = (const float*)d_in[7];
  const float* Wv = (const float*)d_in[8];
  const float* bv = (const float*)d_in[9];
  const float* Wo = (const float*)d_in[10];
  const float* bo = (const float*)d_in[11];
  float* out = (float*)d_out;

  const size_t PER_BUF = (size_t)BATCH * NHEAD * SEQ * HEAD_DIM;  // 4M halves
  half_t* qbuf  = (half_t*)d_ws;
  half_t* kbuf  = qbuf + PER_BUF;
  half_t* vtbuf = kbuf + PER_BUF;   // transposed [B,H,Dh,S]
  half_t* ctx   = vtbuf + PER_BUF;  // [B,S,D] f16

  dim3 blk(256);
  dim3 gemmGrid((BATCH * SEQ) / GT_M, D_MODEL / GT_N);  // 32 x 4
  gemm_qkv_kernel<0><<<gemmGrid, blk, 0, stream>>>(query, Wq, bq, qbuf);
  gemm_qkv_kernel<0><<<gemmGrid, blk, 0, stream>>>(key,   Wk, bk, kbuf);
  gemm_qkv_kernel<1><<<gemmGrid, blk, 0, stream>>>(value, Wv, bv, vtbuf);

  dim3 attnGrid(SEQ / 128, NHEAD, BATCH);  // 16 x 16 x 2
  attn_kernel<<<attnGrid, blk, 0, stream>>>(qbuf, kbuf, vtbuf, ctx);

  gemm_out_kernel<<<gemmGrid, blk, 0, stream>>>(ctx, Wo, bo, out);
}
